// Decoder_75471165325344
// MI455X (gfx1250) — matrix operand order repaired and checked by reading, not compiled, not run on hardware
//
#include <hip/hip_runtime.h>

// ---------------- constants (match reference) ----------------
#define BB     8
#define LATD   512
#define IMGD   64
#define HW     4096          // 64*64
#define NF     64
#define CF     9
#define CIN    576           // NF*CF
#define HIDD   64
#define COUTD  128
#define NDYN   119104
#define NCALLS 8
// dyn vector offsets (elements)
#define OFF_W1 0
#define OFF_B1 36864
#define OFF_W2 36928
#define OFF_B2 45120
#define OFF_WS 45248
#define OFF_BS 118976

typedef __attribute__((ext_vector_type(16))) __bf16 v16bf;
typedef __attribute__((ext_vector_type(8)))  float  v8f;

// ---------------- helpers ----------------
static __device__ __forceinline__ unsigned short f2bf(float f) {
    unsigned int x = __float_as_uint(f);
    unsigned int r = x + 0x7fffu + ((x >> 16) & 1u);   // round-to-nearest-even
    return (unsigned short)(r >> 16);
}

static __device__ __forceinline__ v8f wmma_bf16(v16bf a, v16bf b, v8f c) {
#if defined(__gfx1250__)
    return __builtin_amdgcn_wmma_f32_16x16x32_bf16(false, a, false, b, (short)0, c, false, false);
#else
    return c;
#endif
}

// A fragment (16x32 bf16): lane = (m = l&15, khalf = l>>4)
// VGPR0-7 = K[kbase + 16*khalf .. +16)
static __device__ __forceinline__ v16bf load_a_frag(const unsigned short* rowbase, int kbase, int khalf) {
    const uint4* p0 = (const uint4*)(rowbase + kbase + 16 * khalf);
    const uint4* p1 = (const uint4*)(rowbase + kbase + 16 * khalf + 8);
    union { v16bf v; uint4 q[2]; } u;
    u.q[0] = *p0; u.q[1] = *p1;
    return u.v;
}

// B fragment (32x16 bf16): lane = (n = l&15, khalf = l>>4), holds
// K[kbase + 16*khalf .. +16) of column n, i.e. 32 contiguous bytes of W[o][*].
static __device__ __forceinline__ v16bf load_b_frag(const unsigned short* colbase, int kbase, int khalf) {
    const uint4* p = (const uint4*)(colbase + kbase + 16 * khalf);
    union { v16bf v; uint4 q[2]; } u;
    u.q[0] = p[0]; u.q[1] = p[1];
    return u.v;
}

static __device__ __forceinline__ void threefry2x32(unsigned k0, unsigned k1,
                                                    unsigned x0, unsigned x1,
                                                    unsigned& o0, unsigned& o1) {
    const unsigned rot[8] = {13u, 15u, 26u, 6u, 17u, 29u, 16u, 24u};
    unsigned ks[3]; ks[0] = k0; ks[1] = k1; ks[2] = k0 ^ k1 ^ 0x1BD11BDAu;
    x0 += k0; x1 += k1;
#pragma unroll
    for (int g = 0; g < 5; ++g) {
#pragma unroll
        for (int r = 0; r < 4; ++r) {
            x0 += x1;
            unsigned rr = rot[(g & 1) * 4 + r];
            x1 = (x1 << rr) | (x1 >> (32 - rr));
            x1 ^= x0;
        }
        x0 += ks[(g + 1) % 3];
        x1 += ks[(g + 2) % 3] + (unsigned)(g + 1);
    }
    o0 = x0; o1 = x1;
}

// ---------------- kernel: build gradient-kernel coefficient tables ----------------
// layout: [gx3(9) gy3(9) gx5(25) gy5(25) gx9(81) gy9(81) gx17(289) gy17(289)] = 808 floats
__global__ void k_coef(float* coef) {
    int t = threadIdx.x;
    if (t >= 4) return;
    const int ks[4]   = {3, 5, 9, 17};
    const int offs[4] = {0, 18, 68, 230};
    int k = ks[t], kc = k / 2, off = offs[t];
    float sx = 0.f, sy = 0.f;
    for (int i = 0; i < k; ++i)
        for (int j = 0; j < k; ++j) {
            float x = (float)(j - kc), y = (float)(i - kc);
            float d = x * x + y * y;
            if (i == kc && j == kc) d = 1.0f;
            float gx = x / d, gy = y / d;
            coef[off + i * k + j] = gx;
            coef[off + k * k + i * k + j] = gy;
            sx += fabsf(gx); sy += fabsf(gy);
        }
    for (int i = 0; i < k * k; ++i) {
        coef[off + i] /= sx;
        coef[off + k * k + i] /= sy;
    }
}

// ---------------- kernel: transpose+convert hyp_w (512 x NDYN f32) -> hypT (NDYN x 512 bf16) --
__global__ __launch_bounds__(256) void k_hyp_tr(const float* __restrict__ hw, unsigned short* __restrict__ hypT) {
    __shared__ float tile[32][33];
    int nb = blockIdx.x * 32;   // N base
    int kb = blockIdx.y * 32;   // K base
    int t = threadIdx.x;
    int r0 = t >> 5, cc = t & 31;
#pragma unroll
    for (int i = 0; i < 4; ++i) {
        int r = r0 + 8 * i;
        tile[r][cc] = hw[(size_t)(kb + r) * NDYN + nb + cc];
    }
    __syncthreads();
#pragma unroll
    for (int i = 0; i < 4; ++i) {
        int r = r0 + 8 * i;
        hypT[(size_t)(nb + r) * LATD + kb + cc] = f2bf(tile[cc][r]);
    }
}

// ---------------- kernel: out0 = einsum(seed, sel_w) ----------------
__global__ __launch_bounds__(256) void k_out0(const float* __restrict__ seed, const float* __restrict__ selw,
                                              float* __restrict__ oe0) {
    __shared__ float ssel[NF * 24];
    int b = blockIdx.y;
    int pix = blockIdx.x * 256 + threadIdx.x;
    for (int i = threadIdx.x; i < NF * 24; i += 256) ssel[i] = selw[i];
    __syncthreads();
    float sv[24];
#pragma unroll
    for (int c = 0; c < 24; ++c) sv[c] = seed[(size_t)c * HW + pix];
    float* dst = oe0 + (size_t)b * NF * HW;
    for (int f = 0; f < NF; ++f) {
        float a = 0.f;
#pragma unroll
        for (int c = 0; c < 24; ++c) a += sv[c] * ssel[f * 24 + c];
        dst[(size_t)f * HW + pix] = a;
    }
}

// ---------------- kernel: latent recurrence (one block, 512 threads) ----------------
__global__ __launch_bounds__(512) void k_latent(const float* __restrict__ latin, float* __restrict__ latout,
                                                unsigned short* __restrict__ latbf,
                                                const float* __restrict__ w1, const float* __restrict__ b1,
                                                const float* __restrict__ w2, const float* __restrict__ b2,
                                                const float* __restrict__ ws, const float* __restrict__ bs) {
    __shared__ float slat[BB][LATD];
    __shared__ float sh[BB][LATD];
    int j = threadIdx.x;
#pragma unroll
    for (int i = 0; i < BB; ++i) slat[i][j] = latin[i * LATD + j];
    __syncthreads();
    float acc[BB];
#pragma unroll
    for (int i = 0; i < BB; ++i) acc[i] = b1[j];
    for (int k = 0; k < LATD; ++k) {
        float w = w1[(size_t)k * LATD + j];
#pragma unroll
        for (int i = 0; i < BB; ++i) acc[i] += slat[i][k] * w;
    }
#pragma unroll
    for (int i = 0; i < BB; ++i) sh[i][j] = acc[i] > 0.f ? acc[i] : 0.f;
    __syncthreads();
    float acc2[BB];
    float bb = b2[j] + bs[j];
#pragma unroll
    for (int i = 0; i < BB; ++i) acc2[i] = bb;
    for (int k = 0; k < LATD; ++k) {
        float a = w2[(size_t)k * LATD + j];
        float c = ws[(size_t)k * LATD + j];
#pragma unroll
        for (int i = 0; i < BB; ++i) acc2[i] += sh[i][k] * a + slat[i][k] * c;
    }
#pragma unroll
    for (int i = 0; i < BB; ++i) {
        latout[i * LATD + j] = acc2[i];
        latbf[i * LATD + j] = f2bf(acc2[i]);
    }
#pragma unroll
    for (int i = BB; i < 16; ++i) latbf[i * LATD + j] = 0;   // zero pad rows for WMMA
}

// ---------------- kernel: hypernet GEMM dyn = lat @ hyp_w + hyp_b (WMMA bf16) --------
// A = latbf [16][512] (rows 8..15 zero), B = hypT [NDYN][512]; each wave -> 64 columns
__global__ __launch_bounds__(128) void k_hyper(const unsigned short* __restrict__ latbf,
                                               const unsigned short* __restrict__ hypT,
                                               const float* __restrict__ hypb,
                                               float* __restrict__ dynf, unsigned short* __restrict__ dynb) {
    int gw = blockIdx.x * 4 + (threadIdx.x >> 5);
    if (gw >= NDYN / 64) return;                 // 1861 waves exactly
    int lane = threadIdx.x & 31;
    int n = lane & 15, khalf = lane >> 4;
    int colBase = gw * 64;
    v8f acc[4];
#pragma unroll
    for (int nt = 0; nt < 4; ++nt)
#pragma unroll
        for (int r = 0; r < 8; ++r) acc[nt][r] = 0.f;

    const unsigned short* arow = latbf + (size_t)n * LATD;
    for (int k = 0; k < LATD / 32; ++k) {
        v16bf a = load_a_frag(arow, k * 32, khalf);
#pragma unroll
        for (int nt = 0; nt < 4; ++nt) {
            const unsigned short* brow = hypT + (size_t)(colBase + nt * 16 + n) * LATD;
            __builtin_prefetch(brow + k * 32 + 128, 0, 1);
            v16bf bf = load_b_frag(brow, k * 32, khalf);
            acc[nt] = wmma_bf16(a, bf, acc[nt]);
        }
    }
    if (lane < 16) {                             // rows 8..15 are padding
#pragma unroll
        for (int nt = 0; nt < 4; ++nt) {
            int col = colBase + nt * 16 + lane;  // C layout: lane indexes N
            float hb = hypb[col];
#pragma unroll
            for (int r = 0; r < 8; ++r) {        // VGPR r -> batch r
                float v = acc[nt][r] + hb;
                dynf[(size_t)r * NDYN + col] = v;
                dynb[(size_t)r * NDYN + col] = f2bf(v);
            }
        }
    }
}

// ---------------- kernel: fire mask (JAX threefry: fold_in(key(42), c) -> uniform <= 0.5) ----
__global__ void k_mask(int c, float* __restrict__ maskf) {
    int i = blockIdx.x * blockDim.x + threadIdx.x;
    const int half = (BB * HW) / 2;              // 16384
    if (i >= half) return;
    unsigned f0, f1, o0, o1;
    threefry2x32(0u, 42u, 0u, (unsigned)c, f0, f1);
    threefry2x32(f0, f1, (unsigned)i, (unsigned)(i + half), o0, o1);
    float u0 = __uint_as_float((o0 >> 9) | 0x3f800000u) - 1.0f;
    float u1 = __uint_as_float((o1 >> 9) | 0x3f800000u) - 1.0f;
    maskf[i] = (u0 <= 0.5f) ? 1.0f : 0.0f;
    maskf[i + half] = (u1 <= 0.5f) ? 1.0f : 0.0f;
}

// ---------------- kernel: frac_sobel + instance norm ----------------
// one block per (b, source channel c); produces 9 normalized planes -> percC[b][c+64j][pix] (bf16)
__global__ __launch_bounds__(256) void k_perc(const float* __restrict__ outcur, const float* __restrict__ coef,
                                              unsigned short* __restrict__ percC) {
    __shared__ float simg[HW];
    __shared__ float scoef[808];
    __shared__ float rs1[8], rs2[8], mstat[2];
    int b = blockIdx.y, c = blockIdx.x;
    int t = threadIdx.x, lane = t & 31, wid = t >> 5;
    const float* src = outcur + ((size_t)b * NF + c) * HW;
    for (int i = t; i < HW; i += 256) simg[i] = src[i];
    for (int i = t; i < 808; i += 256) scoef[i] = coef[i];
    __syncthreads();
    const int kk[4] = {3, 5, 9, 17};
    const int dd[4] = {1, 2, 4, 8};
    const int offs[4] = {0, 18, 68, 230};
    for (int j = 0; j < CF; ++j) {
        float vloc[16];
        float s1 = 0.f, s2 = 0.f;
        if (j == 0) {
#pragma unroll
            for (int ii = 0; ii < 16; ++ii) {
                float v = simg[t + 256 * ii];
                vloc[ii] = v; s1 += v; s2 += v * v;
            }
        } else {
            int jj = (j - 1) >> 1;
            int k = kk[jj], d = dd[jj], kc = k >> 1;
            int off = offs[jj] + ((j - 1) & 1) * k * k;
#pragma unroll
            for (int ii = 0; ii < 16; ++ii) {
                int pix = t + 256 * ii;
                int py = pix >> 6, px = pix & 63;
                float v = 0.f;
                for (int ky = 0; ky < k; ++ky) {
                    int yy = py + (ky - kc) * d;
                    if (yy < 0 || yy >= IMGD) continue;
                    for (int kx = 0; kx < k; ++kx) {
                        int xx = px + (kx - kc) * d;
                        if (xx < 0 || xx >= IMGD) continue;
                        v += scoef[off + ky * k + kx] * simg[(yy << 6) | xx];
                    }
                }
                vloc[ii] = v; s1 += v; s2 += v * v;
            }
        }
        for (int o = 16; o; o >>= 1) { s1 += __shfl_down(s1, o); s2 += __shfl_down(s2, o); }
        if (lane == 0) { rs1[wid] = s1; rs2[wid] = s2; }
        __syncthreads();
        if (t == 0) {
            float a = 0.f, q = 0.f;
            for (int i = 0; i < 8; ++i) { a += rs1[i]; q += rs2[i]; }
            float mean = a * (1.0f / HW);
            float var = q * (1.0f / HW) - mean * mean;
            mstat[0] = mean; mstat[1] = rsqrtf(var + 1e-5f);
        }
        __syncthreads();
        float mean = mstat[0], rstd = mstat[1];
        unsigned short* dst = percC + ((size_t)b * CIN + (c + NF * j)) * HW;
#pragma unroll
        for (int ii = 0; ii < 16; ++ii) dst[t + 256 * ii] = f2bf((vloc[ii] - mean) * rstd);
        __syncthreads();
    }
}

// ---------------- kernel: transpose percC [b][CIN][HW] -> percT [b][HW][CIN] (bf16) -------
__global__ __launch_bounds__(256) void k_ptr(const unsigned short* __restrict__ percC,
                                             unsigned short* __restrict__ percT) {
    __shared__ unsigned short tile[32][33];
    int b = blockIdx.z;
    int pb = blockIdx.x * 32;   // pixel base
    int cb = blockIdx.y * 32;   // channel base
    int t = threadIdx.x;
    int r0 = t >> 5, cc = t & 31;
    const unsigned short* src = percC + (size_t)b * CIN * HW;
    unsigned short* dst = percT + (size_t)b * HW * CIN;
#pragma unroll
    for (int i = 0; i < 4; ++i) {
        int r = r0 + 8 * i;
        tile[r][cc] = src[(size_t)(cb + r) * HW + pb + cc];
    }
    __syncthreads();
#pragma unroll
    for (int i = 0; i < 4; ++i) {
        int r = r0 + 8 * i;
        dst[(size_t)(pb + r) * CIN + cb + cc] = tile[cc][r];
    }
}

// ---------------- kernel: per-pixel dynamic MLP + gating (WMMA bf16) ----------------
// block = 256 thr (8 waves); each wave owns 16 pixels; tile of 128 pixels per block.
__global__ __launch_bounds__(256) void k_dyna(const unsigned short* __restrict__ percT,
                                              const float* __restrict__ dynf,
                                              const unsigned short* __restrict__ dynb,
                                              const float* __restrict__ maskf,
                                              float* __restrict__ outNext) {
    __shared__ unsigned short Hs[8][16][HIDD];   // 16 KB
    __shared__ float ytile[8][NF][16];           // 32 KB
    const int b = blockIdx.y;
    const int tile = blockIdx.x;                 // 0..31
    const int wv = threadIdx.x >> 5;
    const int lane = threadIdx.x & 31;
    const int m = lane & 15, khalf = lane >> 4;
    const int row0 = tile * 128 + wv * 16;

    const unsigned short* Ab = percT + (size_t)b * HW * CIN;
    const unsigned short* W1 = dynb + (size_t)b * NDYN + OFF_W1;
    const unsigned short* W2 = dynb + (size_t)b * NDYN + OFF_W2;
    const unsigned short* WS = dynb + (size_t)b * NDYN + OFF_WS;
    const float* B1 = dynf + (size_t)b * NDYN + OFF_B1;
    const float* B2 = dynf + (size_t)b * NDYN + OFF_B2;
    const float* BS = dynf + (size_t)b * NDYN + OFF_BS;

    const unsigned short* arow = Ab + (size_t)(row0 + m) * CIN;

    // ---- layer 1: H = relu(perc @ W1^T + b1), 16x64 ----
    v8f acc1[4];
#pragma unroll
    for (int nt = 0; nt < 4; ++nt)
#pragma unroll
        for (int r = 0; r < 8; ++r) acc1[nt][r] = 0.f;
    for (int k = 0; k < CIN / 32; ++k) {
        v16bf a = load_a_frag(arow, k * 32, khalf);
#pragma unroll
        for (int nt = 0; nt < 4; ++nt) {
            v16bf bf = load_b_frag(W1 + (size_t)(nt * 16 + m) * CIN, k * 32, khalf);
            acc1[nt] = wmma_bf16(a, bf, acc1[nt]);
        }
    }
#pragma unroll
    for (int nt = 0; nt < 4; ++nt) {
        int o = nt * 16 + m;                     // lane's column (C layout: lane = N)
        float bias = B1[o];
#pragma unroll
        for (int r = 0; r < 8; ++r) {
            int row = r + 8 * khalf;
            float h = acc1[nt][r] + bias;
            Hs[wv][row][o] = f2bf(h > 0.f ? h : 0.f);
        }
    }
    __syncthreads();

    // ---- layer 2 + skip: Y = H @ W2^T + perc @ WS^T (+biases), 16x128 ----
    v8f acc2[8];
#pragma unroll
    for (int nt = 0; nt < 8; ++nt)
#pragma unroll
        for (int r = 0; r < 8; ++r) acc2[nt][r] = 0.f;
#pragma unroll
    for (int k = 0; k < HIDD / 32; ++k) {
        v16bf a = load_a_frag(&Hs[wv][m][0], k * 32, khalf);
#pragma unroll
        for (int nt = 0; nt < 8; ++nt) {
            v16bf bf = load_b_frag(W2 + (size_t)(nt * 16 + m) * HIDD, k * 32, khalf);
            acc2[nt] = wmma_bf16(a, bf, acc2[nt]);
        }
    }
    for (int k = 0; k < CIN / 32; ++k) {
        v16bf a = load_a_frag(arow, k * 32, khalf);
#pragma unroll
        for (int nt = 0; nt < 8; ++nt) {
            v16bf bf = load_b_frag(WS + (size_t)(nt * 16 + m) * CIN, k * 32, khalf);
            acc2[nt] = wmma_bf16(a, bf, acc2[nt]);
        }
    }

    // ---- gating epilogue: y = (val+b) * sigmoid(gate+b) * mask ----
#pragma unroll
    for (int nt = 0; nt < 4; ++nt) {
        int o = nt * 16 + m;
        int og = o + NF;
        float bv = B2[o] + BS[o];
        float bg = B2[og] + BS[og];
#pragma unroll
        for (int r = 0; r < 8; ++r) {
            int row = r + 8 * khalf;
            float val = acc2[nt][r] + bv;
            float gate = acc2[nt + 4][r] + bg;
            float msk = maskf[b * HW + row0 + row];
            float sig = 1.0f / (1.0f + __expf(-gate));
            ytile[wv][o][row] = val * sig * msk;
        }
    }
    // coalesced channel-major store via LDS (wave-local)
    float* dst = outNext + (size_t)b * NF * HW;
    for (int idx = lane; idx < NF * 16; idx += 32) {
        int col = idx >> 4, row = idx & 15;
        dst[(size_t)col * HW + row0 + row] = ytile[wv][col][row];
    }
}

// ---------------- kernel: final 64->3 projection + clip ----------------
__global__ __launch_bounds__(256) void k_final(const float* __restrict__ outE, const float* __restrict__ ocw,
                                               const float* __restrict__ ocb,
                                               float* __restrict__ outClip, float* __restrict__ outRaw) {
    __shared__ float sw[3 * NF];
    int b = blockIdx.y;
    int pix = blockIdx.x * 256 + threadIdx.x;
    for (int i = threadIdx.x; i < 3 * NF; i += 256) sw[i] = ocw[i];
    __syncthreads();
    const float* src = outE + (size_t)b * NF * HW;
    float a0 = ocb[0], a1 = ocb[1], a2 = ocb[2];
    for (int c = 0; c < NF; ++c) {
        float v = src[(size_t)c * HW + pix];
        a0 += v * sw[0 * NF + c];
        a1 += v * sw[1 * NF + c];
        a2 += v * sw[2 * NF + c];
    }
    size_t o0 = ((size_t)b * 3 + 0) * HW + pix;
    size_t o1 = ((size_t)b * 3 + 1) * HW + pix;
    size_t o2 = ((size_t)b * 3 + 2) * HW + pix;
    outRaw[o0] = a0; outRaw[o1] = a1; outRaw[o2] = a2;
    outClip[o0] = fminf(fmaxf(a0, -1.f), 1.f);
    outClip[o1] = fminf(fmaxf(a1, -1.f), 1.f);
    outClip[o2] = fminf(fmaxf(a2, -1.f), 1.f);
}

// ---------------- host launcher ----------------
extern "C" void kernel_launch(void* const* d_in, const int* in_sizes, int n_in,
                              void* d_out, int out_size, void* d_ws, size_t ws_size,
                              hipStream_t stream) {
    const float* lat   = (const float*)d_in[0];
    const float* seed  = (const float*)d_in[1];
    const float* selw  = (const float*)d_in[2];
    const float* flw1  = (const float*)d_in[3];
    const float* flb1  = (const float*)d_in[4];
    const float* flw2  = (const float*)d_in[5];
    const float* flb2  = (const float*)d_in[6];
    const float* flws  = (const float*)d_in[7];
    const float* flbs  = (const float*)d_in[8];
    const float* hypw  = (const float*)d_in[9];
    const float* hypb  = (const float*)d_in[10];
    const float* ocw   = (const float*)d_in[11];
    const float* ocb   = (const float*)d_in[12];
    float* outF = (float*)d_out;

    // workspace carve (256B aligned)
    char* w = (char*)d_ws;
    auto alloc = [&](size_t bytes) -> void* {
        void* p = (void*)w;
        w += (bytes + 255) & ~(size_t)255;
        return p;
    };
    unsigned short* hypT  = (unsigned short*)alloc((size_t)NDYN * LATD * 2);
    unsigned short* percC = (unsigned short*)alloc((size_t)BB * CIN * HW * 2);
    unsigned short* percT = (unsigned short*)alloc((size_t)BB * CIN * HW * 2);
    float*          dynf  = (float*)alloc((size_t)BB * NDYN * 4);
    unsigned short* dynb  = (unsigned short*)alloc((size_t)BB * NDYN * 2);
    float*          lat0  = (float*)alloc((size_t)BB * LATD * 4);
    float*          lat1  = (float*)alloc((size_t)BB * LATD * 4);
    unsigned short* latbf = (unsigned short*)alloc((size_t)16 * LATD * 2);
    float*          maskf = (float*)alloc((size_t)BB * HW * 4);
    float*          coef  = (float*)alloc((size_t)808 * 4);

    const size_t stepSz = (size_t)BB * NF * HW;              // 2,097,152 floats
    auto OE = [&](int s) -> float* { return outF + 98304 + (size_t)s * stepSz; };

    k_coef<<<1, 64, 0, stream>>>(coef);
    k_hyp_tr<<<dim3(NDYN / 32, LATD / 32), 256, 0, stream>>>(hypw, hypT);
    k_out0<<<dim3(HW / 256, BB), 256, 0, stream>>>(seed, selw, OE(0));

    const float* curLat = lat;
    for (int c = 0; c < NCALLS; ++c) {
        float* nl = (c & 1) ? lat1 : lat0;
        k_latent<<<1, 512, 0, stream>>>(curLat, nl, latbf, flw1, flb1, flw2, flb2, flws, flbs);
        curLat = nl;
        k_hyper<<<(NDYN / 64 + 3) / 4, 128, 0, stream>>>(latbf, hypT, hypb, dynf, dynb);
        k_mask<<<(BB * HW / 2) / 256, 256, 0, stream>>>(c, maskf);
        k_perc<<<dim3(NF, BB), 256, 0, stream>>>(OE(c), coef, percC);
        k_ptr<<<dim3(HW / 32, CIN / 32, BB), 256, 0, stream>>>(percC, percT);
        k_dyna<<<dim3(HW / 128, BB), 256, 0, stream>>>(percT, dynf, dynb, maskf, OE(c + 1));
    }
    k_final<<<dim3(HW / 256, BB), 256, 0, stream>>>(OE(NCALLS), ocw, ocb,
                                                    outF, outF + 98304 + 9 * stepSz);
}
